// NeuralVSR_64819646431843
// MI455X (gfx1250) — matrix operand order repaired and checked by reading, not compiled, not run on hardware
//
#include <hip/hip_runtime.h>

// ---------------------------------------------------------------- constants
constexpr int   N_NODES = 50000;
constexpr int   E_EDGES = 800000;
constexpr int   M_EDGES = E_EDGES + N_NODES;   // + self loops
constexpr int   HID     = 128;
constexpr int   HEADS   = 4;
constexpr int   CH      = 32;                  // HID / HEADS
constexpr int   NLAYERS = 3;
constexpr float SLOPE   = 0.2f;
constexpr float LN_EPS  = 1e-5f;

typedef __attribute__((ext_vector_type(16))) __bf16 v16bf;
typedef __attribute__((ext_vector_type(8)))  float  v8f;

union Frag16 { v16bf v; uint4 q[2]; };

static __device__ __forceinline__ unsigned short f2bf(float f) {
  unsigned u = __float_as_uint(f);
  u += 0x7FFFu + ((u >> 16) & 1u);             // round to nearest even
  return (unsigned short)(u >> 16);
}

// order-preserving float -> uint mapping for atomicMax on signed floats
static __device__ __forceinline__ unsigned fmap(float f) {
  unsigned b = __float_as_uint(f);
  return (b & 0x80000000u) ? ~b : (b | 0x80000000u);
}
static __device__ __forceinline__ float funmap(unsigned u) {
  unsigned b = (u & 0x80000000u) ? (u ^ 0x80000000u) : ~u;
  return __uint_as_float(b);
}

static __device__ __forceinline__ float wave_sum(float v) {
  #pragma unroll
  for (int off = 16; off > 0; off >>= 1) v += __shfl_xor(v, off, 32);
  return v;
}

// ---------------------------------------------------------------- utility
__global__ void clear_f32_kernel(float* __restrict__ p, int n) {
  int i = blockIdx.x * blockDim.x + threadIdx.x;
  if (i < n) p[i] = 0.0f;
}

// mean of edge_attr columns (EF = 4): partial wave sums -> atomicAdd
__global__ void ea_reduce_kernel(const float* __restrict__ ea,
                                 float* __restrict__ sum4) {
  int t = blockIdx.x * blockDim.x + threadIdx.x;
  int stride = gridDim.x * blockDim.x;
  float s0 = 0.f, s1 = 0.f, s2 = 0.f, s3 = 0.f;
  for (int i = t; i < E_EDGES; i += stride) {
    const float* p = ea + (size_t)i * 4;
    s0 += p[0]; s1 += p[1]; s2 += p[2]; s3 += p[3];
  }
  s0 = wave_sum(s0); s1 = wave_sum(s1); s2 = wave_sum(s2); s3 = wave_sum(s3);
  if ((threadIdx.x & 31) == 0) {
    atomicAdd(&sum4[0], s0); atomicAdd(&sum4[1], s1);
    atomicAdd(&sum4[2], s2); atomicAdd(&sum4[3], s3);
  }
}

// ------------------------------------------------- weight pack for WMMA B
// Pack the six 128x128 f32 matrices (lin_l_w[0..2], lin_r_w[0..2]) into the
// gfx1250 WMMA bf16 B-fragment layout (16-bit B 32x16, wave32):
//   lane 0-15  : column N = lane,    K = kt*32 + 0..7, 16..23  (elems 0..15)
//   lane 16-31 : column N = lane-16, K = kt*32 + 8..15, 24..31
// Packed index: (((mat*8 + ct)*4 + kt)*32 + lane)*16 + e
__global__ void pack_weights_kernel(const float* __restrict__ lw,
                                    const float* __restrict__ rw,
                                    unsigned short* __restrict__ wp) {
  int idx = blockIdx.x * blockDim.x + threadIdx.x;
  if (idx >= 6 * 16384) return;
  int mat  = idx / 16384;
  int r    = idx & 16383;
  int ct   = r >> 11;            // 0..7  column tile
  int kt   = (r >> 9) & 3;       // 0..3  k tile
  int lane = (r >> 4) & 31;
  int e    = r & 15;
  int K  = kt * 32 + ((lane >> 4) << 3) + (e & 7) + ((e >> 3) << 4);
  int Nc = ct * 16 + (lane & 15);
  const float* W = (mat < 3) ? (lw + (size_t)mat * 16384)
                             : (rw + (size_t)(mat - 3) * 16384);
  wp[idx] = f2bf(W[(size_t)K * HID + Nc]);
}

// ------------------------------------------------------ input projection
// h = relu([centers, node_features] @ in_w + in_b), also emit bf16 copy
__global__ void input_proj_kernel(const float* __restrict__ centers,
                                  const float* __restrict__ nf,
                                  const float* __restrict__ in_w,
                                  const float* __restrict__ in_b,
                                  float* __restrict__ h,
                                  unsigned short* __restrict__ hb) {
  int idx = blockIdx.x * blockDim.x + threadIdx.x;
  if (idx >= N_NODES * HID) return;
  int n = idx >> 7, c = idx & 127;
  float s = in_b[c];
  s += centers[n * 2 + 0] * in_w[0 * HID + c];
  s += centers[n * 2 + 1] * in_w[1 * HID + c];
  #pragma unroll
  for (int j = 0; j < 5; ++j)
    s += nf[n * 5 + j] * in_w[(2 + j) * HID + c];
  s = fmaxf(s, 0.0f);
  h[idx]  = s;
  hb[idx] = f2bf(s);
}

// ------------------------------------------------------ fused WMMA GEMM
// Computes BOTH  xl = Hb @ Wl + bl  and  xr = Hb @ Wr + br  in one pass:
// the A fragments (activations) are identical for the two products, so each
// wave loads A once per k-step and feeds two v_wmma chains (8 WMMAs/wave).
// block = 256 threads = 8 waves; wave w owns column tile w; block = 16 rows.
__global__ void gemm128_dual_wmma_kernel(const unsigned short* __restrict__ hb,
                                         const unsigned short* __restrict__ wpl,
                                         const unsigned short* __restrict__ wpr,
                                         const float* __restrict__ bl,
                                         const float* __restrict__ br,
                                         float* __restrict__ yl,
                                         float* __restrict__ yr) {
  int ct    = threadIdx.x >> 5;        // 0..7
  int lane  = threadIdx.x & 31;
  int tileM = blockIdx.x * 16;
  int m     = lane & 15;
  int lhalf = lane >> 4;               // 0 or 1
  const unsigned short* hrow = hb + (size_t)(tileM + m) * HID;

  v8f accL = {};
  v8f accR = {};
  #pragma unroll
  for (int kt = 0; kt < 4; ++kt) {
    // A fragment: 16-bit A 16x32 lane map (two contiguous 16B chunks)
    Frag16 a;
    int kb = kt * 32 + lhalf * 8;
    a.q[0] = *reinterpret_cast<const uint4*>(hrow + kb);
    a.q[1] = *reinterpret_cast<const uint4*>(hrow + kb + 16);
    // B fragments: prepacked, 32B contiguous per lane
    size_t boff = ((size_t)(ct * 4 + kt) * 32 + lane) << 4;
    Frag16 b0, b1;
    b0.q[0] = reinterpret_cast<const uint4*>(wpl + boff)[0];
    b0.q[1] = reinterpret_cast<const uint4*>(wpl + boff)[1];
    b1.q[0] = reinterpret_cast<const uint4*>(wpr + boff)[0];
    b1.q[1] = reinterpret_cast<const uint4*>(wpr + boff)[1];
    accL = __builtin_amdgcn_wmma_f32_16x16x32_bf16(
        false, a.v, false, b0.v, (short)0, accL, false, false);
    accR = __builtin_amdgcn_wmma_f32_16x16x32_bf16(
        false, a.v, false, b1.v, (short)0, accR, false, false);
  }

  // C/D layout: VGPR j, lane -> M = j + (lane>=16)*8, N = lane&15
  int ncol  = ct * 16 + (lane & 15);
  int mbase = tileM + lhalf * 8;
  float bvL = bl[ncol];
  float bvR = br[ncol];
  #pragma unroll
  for (int j = 0; j < 8; ++j) {
    yl[(size_t)(mbase + j) * HID + ncol] = accL[j] + bvL;
    yr[(size_t)(mbase + j) * HID + ncol] = accR[j] + bvR;
  }
}

// --------------------------------------------- edge pass 1: logits + max
// one wave per edge; lane l covers channel l of every head (c = h*32 + l)
__global__ void edge_logits_kernel(const int* __restrict__ srcp,
                                   const int* __restrict__ dstp,
                                   const float* __restrict__ eattr,
                                   const float* __restrict__ ea_sum,
                                   const float* __restrict__ xl,
                                   const float* __restrict__ xr,
                                   const float* __restrict__ We,   // 4 x 128
                                   const float* __restrict__ att,  // 4 x 32
                                   float* __restrict__ logits,
                                   unsigned* __restrict__ lmax,
                                   float invE) {
  __shared__ float sWe[4 * HID];
  __shared__ float sAtt[HID];
  int t = threadIdx.x;
  for (int j = t; j < 4 * HID; j += blockDim.x) sWe[j] = We[j];
  for (int j = t; j < HID;     j += blockDim.x) sAtt[j] = att[j];
  __syncthreads();

  int wave = t >> 5, lane = t & 31;
  int edge = blockIdx.x * 8 + wave;
  if (edge >= M_EDGES) return;

  int s, d; float ea0, ea1, ea2, ea3;
  if (edge < E_EDGES) {
    s = srcp[edge]; d = dstp[edge];
    const float* p = eattr + (size_t)edge * 4;
    ea0 = p[0]; ea1 = p[1]; ea2 = p[2]; ea3 = p[3];
  } else {                                   // self loop, mean edge_attr
    s = d = edge - E_EDGES;
    ea0 = ea_sum[0] * invE; ea1 = ea_sum[1] * invE;
    ea2 = ea_sum[2] * invE; ea3 = ea_sum[3] * invE;
  }

  float lg[HEADS];
  #pragma unroll
  for (int h = 0; h < HEADS; ++h) {
    int c = h * CH + lane;
    float ev = ea0 * sWe[c] + ea1 * sWe[HID + c]
             + ea2 * sWe[2 * HID + c] + ea3 * sWe[3 * HID + c];
    float m = xl[(size_t)s * HID + c] + xr[(size_t)d * HID + c] + ev;
    m = (m > 0.0f) ? m : SLOPE * m;          // GATv2 leaky_relu
    lg[h] = wave_sum(m * sAtt[c]);
  }
  if (lane == 0) {
    #pragma unroll
    for (int h = 0; h < HEADS; ++h) {
      logits[(size_t)edge * HEADS + h] = lg[h];
      atomicMax(&lmax[(size_t)d * HEADS + h], fmap(lg[h]));
    }
  }
}

// -------------------------------- edge pass 2: exp, denom, weighted scatter
__global__ void edge_accum_kernel(const int* __restrict__ srcp,
                                  const int* __restrict__ dstp,
                                  const float* __restrict__ xl,
                                  const float* __restrict__ logits,
                                  const unsigned* __restrict__ lmax,
                                  float* __restrict__ den,
                                  float* __restrict__ agg) {
  int t = threadIdx.x;
  int wave = t >> 5, lane = t & 31;
  int edge = blockIdx.x * 8 + wave;
  if (edge >= M_EDGES) return;

  int s, d;
  if (edge < E_EDGES) { s = srcp[edge]; d = dstp[edge]; }
  else                { s = d = edge - E_EDGES; }

  float ex[HEADS];
  #pragma unroll
  for (int h = 0; h < HEADS; ++h)
    ex[h] = __expf(logits[(size_t)edge * HEADS + h] -
                   funmap(lmax[(size_t)d * HEADS + h]));
  if (lane == 0) {
    #pragma unroll
    for (int h = 0; h < HEADS; ++h)
      atomicAdd(&den[(size_t)d * HEADS + h], ex[h]);
  }
  #pragma unroll
  for (int h = 0; h < HEADS; ++h) {
    int c = h * CH + lane;
    atomicAdd(&agg[(size_t)d * HID + c],
              ex[h] * xl[(size_t)s * HID + c]);
  }
}

// ----------------- node epilogue: normalize, bias, relu, residual, LayerNorm
__global__ void node_post_kernel(const float* __restrict__ agg,
                                 const float* __restrict__ den,
                                 const float* __restrict__ conv_b,
                                 const float* __restrict__ ln_g,
                                 const float* __restrict__ ln_b,
                                 float* __restrict__ h,
                                 unsigned short* __restrict__ hb) {
  __shared__ float wred[4];
  int n = blockIdx.x;
  int t = threadIdx.x;                      // 0..127, 4 waves
  float a = agg[(size_t)n * HID + t] / den[(size_t)n * HEADS + (t >> 5)];
  float g = fmaxf(a + conv_b[t], 0.0f);     // relu(agg + conv_b)
  float x = g + h[(size_t)n * HID + t];     // residual

  float s = wave_sum(x);
  if ((t & 31) == 0) wred[t >> 5] = s;
  __syncthreads();
  float mu = (wred[0] + wred[1] + wred[2] + wred[3]) * (1.0f / HID);
  __syncthreads();

  float xc = x - mu;
  s = wave_sum(xc * xc);
  if ((t & 31) == 0) wred[t >> 5] = s;
  __syncthreads();
  float var = (wred[0] + wred[1] + wred[2] + wred[3]) * (1.0f / HID);

  float y = xc * rsqrtf(var + LN_EPS) * ln_g[t] + ln_b[t];
  h[(size_t)n * HID + t]  = y;
  hb[(size_t)n * HID + t] = f2bf(y);
}

// ----------------------------------------------------- output projection
__global__ void out_proj_kernel(const float* __restrict__ h,
                                const float* __restrict__ out_w,  // 128 x 2
                                const float* __restrict__ out_b,
                                float* __restrict__ out) {
  int wave = threadIdx.x >> 5, lane = threadIdx.x & 31;
  int n = blockIdx.x * 8 + wave;
  if (n >= N_NODES) return;
  float s0 = 0.f, s1 = 0.f;
  #pragma unroll
  for (int kk = 0; kk < 4; ++kk) {
    int k = lane + kk * 32;
    float hv = h[(size_t)n * HID + k];
    s0 += hv * out_w[k * 2 + 0];
    s1 += hv * out_w[k * 2 + 1];
  }
  s0 = wave_sum(s0);
  s1 = wave_sum(s1);
  if (lane == 0) {
    out[n * 2 + 0] = s0 + out_b[0];
    out[n * 2 + 1] = s1 + out_b[1];
  }
}

// ---------------------------------------------------------------- launcher
static inline int cdiv(long long a, long long b) { return (int)((a + b - 1) / b); }

extern "C" void kernel_launch(void* const* d_in, const int* in_sizes, int n_in,
                              void* d_out, int out_size, void* d_ws, size_t ws_size,
                              hipStream_t stream) {
  (void)in_sizes; (void)n_in; (void)out_size; (void)ws_size;

  const float* centers  = (const float*)d_in[0];
  const float* node_ft  = (const float*)d_in[1];
  const int*   eidx     = (const int*)  d_in[2];   // (2, E): src then dst
  const float* eattr    = (const float*)d_in[3];
  const float* in_w     = (const float*)d_in[4];
  const float* in_b     = (const float*)d_in[5];
  const float* lin_l_w  = (const float*)d_in[6];
  const float* lin_l_b  = (const float*)d_in[7];
  const float* lin_r_w  = (const float*)d_in[8];
  const float* lin_r_b  = (const float*)d_in[9];
  const float* lin_e_w  = (const float*)d_in[10];
  const float* att_w    = (const float*)d_in[11];
  const float* conv_b   = (const float*)d_in[12];
  const float* ln_g     = (const float*)d_in[13];
  const float* ln_b     = (const float*)d_in[14];
  const float* out_w    = (const float*)d_in[15];
  const float* out_b    = (const float*)d_in[16];

  const int* srcp = eidx;
  const int* dstp = eidx + E_EDGES;

  // ---- workspace carve-out (256B aligned) ----
  char* ws = (char*)d_ws;
  size_t off = 0;
  auto carve = [&](size_t bytes) -> void* {
    void* p = ws + off;
    off = (off + bytes + 255) & ~(size_t)255;
    return p;
  };
  float*          h      = (float*)         carve((size_t)N_NODES * HID * 4);
  unsigned short* hb     = (unsigned short*)carve((size_t)N_NODES * HID * 2);
  float*          xl     = (float*)         carve((size_t)N_NODES * HID * 4);
  float*          xr     = (float*)         carve((size_t)N_NODES * HID * 4);
  // agg | den | lmax contiguous so one clear covers all three
  float*          agg    = (float*)         carve((size_t)N_NODES * (HID + 2 * HEADS) * 4);
  float*          den    = agg + (size_t)N_NODES * HID;
  unsigned*       lmax   = (unsigned*)(den + (size_t)N_NODES * HEADS);
  float*          logits = (float*)         carve((size_t)M_EDGES * HEADS * 4);
  unsigned short* wp     = (unsigned short*)carve((size_t)6 * 16384 * 2);
  float*          ea_sum = (float*)         carve(4 * 4);

  const int clearN = N_NODES * (HID + 2 * HEADS);   // agg + den + lmax words
  const float invE = 1.0f / (float)E_EDGES;

  // ---- one-time per launch (deterministic) ----
  pack_weights_kernel<<<cdiv(6 * 16384, 256), 256, 0, stream>>>(lin_l_w, lin_r_w, wp);
  clear_f32_kernel<<<1, 32, 0, stream>>>(ea_sum, 4);
  ea_reduce_kernel<<<512, 256, 0, stream>>>(eattr, ea_sum);
  input_proj_kernel<<<cdiv((long long)N_NODES * HID, 256), 256, 0, stream>>>(
      centers, node_ft, in_w, in_b, h, hb);

  // ---- GAT layers ----
  for (int i = 0; i < NLAYERS; ++i) {
    gemm128_dual_wmma_kernel<<<N_NODES / 16, 256, 0, stream>>>(
        hb, wp + (size_t)i * 16384, wp + (size_t)(3 + i) * 16384,
        lin_l_b + (size_t)i * HID, lin_r_b + (size_t)i * HID, xl, xr);
    clear_f32_kernel<<<cdiv(clearN, 256), 256, 0, stream>>>(agg, clearN);
    edge_logits_kernel<<<cdiv(M_EDGES, 8), 256, 0, stream>>>(
        srcp, dstp, eattr, ea_sum, xl, xr,
        lin_e_w + (size_t)i * 4 * HID, att_w + (size_t)i * HID,
        logits, lmax, invE);
    edge_accum_kernel<<<cdiv(M_EDGES, 8), 256, 0, stream>>>(
        srcp, dstp, xl, logits, lmax, den, agg);
    node_post_kernel<<<N_NODES, HID, 0, stream>>>(
        agg, den, conv_b + (size_t)i * HID,
        ln_g + (size_t)i * HID, ln_b + (size_t)i * HID, h, hb);
  }

  out_proj_kernel<<<cdiv(N_NODES, 8), 256, 0, stream>>>(h, out_w, out_b, (float*)d_out);
}
